// DeepseekECMoE_70875550319382
// MI455X (gfx1250) — matrix operand order repaired and checked by reading, not compiled, not run on hardware
//
#include <hip/hip_runtime.h>

// ---------------------------------------------------------------------------
// DeepSeek expert-choice MoE forward for MI455X (gfx1250), wave32 + WMMA bf16.
// Compute-bound (~310 GFLOP vs ~290MB unique traffic) -> all GEMMs run through
// v_wmma_f32_16x16x32_bf16 with fp32 accumulation. GEMM tiles are staged into
// LDS with GLOBAL_LOAD_ASYNC_TO_LDS_B128 (ASYNCcnt) and double-buffered.
// ---------------------------------------------------------------------------

typedef unsigned short u16;
typedef unsigned int   u32;
typedef __attribute__((ext_vector_type(16))) __bf16 v16bf;
typedef __attribute__((ext_vector_type(8)))  float  v8f;
typedef __attribute__((ext_vector_type(4)))  int    v4i;

typedef __attribute__((address_space(3))) void lds_void;
typedef __attribute__((address_space(1))) void glob_void;
typedef __attribute__((address_space(1))) v4i  glob_v4i;
typedef __attribute__((address_space(3))) v4i  lds_v4i;

constexpr int Bb   = 8;
constexpr int Ss   = 1024;
constexpr int Hh   = 1024;
constexpr int Ee   = 8;
constexpr int Ii   = 2048;
constexpr int Ish  = 2048;
constexpr int CAPc = 256;

__device__ __forceinline__ u16 f2bf(float f) {
  u32 u = __float_as_uint(f);
  u32 r = u + 0x7FFFu + ((u >> 16) & 1u);   // round-to-nearest-even
  return (u16)(r >> 16);
}

__device__ __forceinline__ float gelu_exact(float x) {
  return 0.5f * x * (1.0f + erff(x * 0.70710678118654752f));
}

__device__ __forceinline__ v8f wmma_bf16(v16bf a, v16bf b, v8f c) {
  return __builtin_amdgcn_wmma_f32_16x16x32_bf16(false, a, false, b,
                                                 (short)0, c, false, false);
}

// ---- async copy 16B global -> LDS (ASYNCcnt-tracked, no VGPR data path) ----
__device__ __forceinline__ void async_cp16(const u16* gsrc, u16* ldst) {
#if __has_builtin(__builtin_amdgcn_global_load_async_to_lds_b128)
  __builtin_amdgcn_global_load_async_to_lds_b128(
      (glob_v4i*)(glob_void*)gsrc, (lds_v4i*)(lds_void*)ldst, 0, 0);
#else
  unsigned loff = (unsigned)(unsigned long long)(lds_void*)ldst;
  asm volatile("global_load_async_to_lds_b128 %0, %1, off"
               :: "v"(loff), "v"(gsrc) : "memory");
#endif
}

template <int N>
__device__ __forceinline__ void wait_async() {
#if __has_builtin(__builtin_amdgcn_s_wait_asynccnt)
  __builtin_amdgcn_s_wait_asynccnt(N);
#else
  asm volatile("s_wait_asynccnt %0" :: "i"(N) : "memory");
#endif
}

// A fragment: 16x32 (MxK), source row-major [rows, K] bf16.
__device__ __forceinline__ v16bf load_fragA(const u16* base, int ld,
                                            int row0, int k0, int lane) {
  int hf = lane >> 4, r = lane & 15;
  const u16* p = base + (size_t)(row0 + r) * ld + k0 + hf * 16;
  union { uint4 q[2]; v16bf v; } u;
  u.q[0] = *(const uint4*)p;
  u.q[1] = *(const uint4*)(p + 8);
  return u.v;
}

// B fragment: 32x16 (KxN), weights stored row-major [N, K] bf16.
__device__ __forceinline__ v16bf load_fragB_bf16(const u16* base, int ld,
                                                 int col0, int k0, int lane) {
  int hf = lane >> 4, n = lane & 15;
  const u16* p = base + (size_t)(col0 + n) * ld + k0 + hf * 16;
  union { uint4 q[2]; v16bf v; } u;
  u.q[0] = *(const uint4*)p;
  u.q[1] = *(const uint4*)(p + 8);
  return u.v;
}

// fp32 weights -> bf16 fragments on the fly (fallback path)
__device__ __forceinline__ v16bf load_fragB_f32(const float* base, int ld,
                                                int col0, int k0, int lane) {
  int hf = lane >> 4, n = lane & 15;
  const float* p = base + (size_t)(col0 + n) * ld + k0 + hf * 16;
  union { u16 s[16]; v16bf v; } u;
#pragma unroll
  for (int i = 0; i < 16; i += 4) {
    float4 f = *(const float4*)(p + i);
    u.s[i + 0] = f2bf(f.x);
    u.s[i + 1] = f2bf(f.y);
    u.s[i + 2] = f2bf(f.z);
    u.s[i + 3] = f2bf(f.w);
  }
  return u.v;
}

template <bool WF32>
__device__ __forceinline__ v16bf load_fragW(const void* base, int ld,
                                            int col0, int k0, int lane) {
  if constexpr (WF32)
    return load_fragB_f32((const float*)base, ld, col0, k0, lane);
  else
    return load_fragB_bf16((const u16*)base, ld, col0, k0, lane);
}

// --------------------------- small helper kernels ---------------------------

__global__ void k_cvt(const float* __restrict__ src, u16* __restrict__ dst, int n) {
  int i = (blockIdx.x * blockDim.x + threadIdx.x) * 4;
  if (i >= n) return;
  float4 f = *(const float4*)(src + i);
  union { u16 s[4]; uint2 u; } o;
  o.s[0] = f2bf(f.x); o.s[1] = f2bf(f.y); o.s[2] = f2bf(f.z); o.s[3] = f2bf(f.w);
  *(uint2*)(dst + i) = o.u;
}

// one wave32 per token: logits over E=8, softmax, write affinity [b][e][s]
__global__ void k_gating(const float* __restrict__ hs, const float* __restrict__ gw,
                         float* __restrict__ aff) {
  int lane  = threadIdx.x & 31;
  int token = blockIdx.x * (blockDim.x >> 5) + (threadIdx.x >> 5);
  const float* x = hs + (size_t)token * Hh;
  float acc[Ee];
#pragma unroll
  for (int e = 0; e < Ee; e++) acc[e] = 0.0f;
  for (int h = lane; h < Hh; h += 32) {
    float xv = x[h];
    const float* g = gw + h * Ee;
#pragma unroll
    for (int e = 0; e < Ee; e++) acc[e] += xv * g[e];
  }
#pragma unroll
  for (int e = 0; e < Ee; e++)
    for (int off = 16; off > 0; off >>= 1)
      acc[e] += __shfl_xor(acc[e], off, 32);
  if (lane == 0) {
    float m = acc[0];
#pragma unroll
    for (int e = 1; e < Ee; e++) m = fmaxf(m, acc[e]);
    float ex[Ee];
    float s = 0.0f;
#pragma unroll
    for (int e = 0; e < Ee; e++) { ex[e] = expf(acc[e] - m); s += ex[e]; }
    float inv = 1.0f / s;
    int b = token >> 10, sr = token & (Ss - 1);
#pragma unroll
    for (int e = 0; e < Ee; e++)
      aff[((size_t)b * Ee + e) * Ss + sr] = ex[e] * inv;
  }
}

// top-CAP of S per (b,e) via stable rank-by-counting (matches lax.top_k set)
__global__ void k_topk(const float* __restrict__ aff, int* __restrict__ idx,
                       float* __restrict__ score) {
  __shared__ float a[Ss];
  int g = blockIdx.x;                  // b*E + e
  const float* ap = aff + (size_t)g * Ss;
  for (int s = threadIdx.x; s < Ss; s += blockDim.x) a[s] = ap[s];
  __syncthreads();
  for (int s = threadIdx.x; s < Ss; s += blockDim.x) {
    float v = a[s];
    int rank = 0;
    for (int j = 0; j < Ss; j++) {
      float w = a[j];
      rank += ((w > v) || (w == v && j < s)) ? 1 : 0;
    }
    if (rank < CAPc) {
      idx[g * CAPc + rank]   = s;
      score[g * CAPc + rank] = v;
    }
  }
}

// gather selected tokens (bf16) into dense [b*E+e][cap][H]
__global__ void k_gather(const u16* __restrict__ Xs, const int* __restrict__ idx,
                         u16* __restrict__ Xr) {
  int slot = blockIdx.x;
  int g = slot / CAPc;
  int b = g / Ee;
  int tok = idx[slot];
  const u16* src = Xs + ((size_t)b * Ss + tok) * Hh;
  u16* dst = Xr + (size_t)slot * Hh;
  int t = threadIdx.x;                 // 128 threads * 16B = 2KB row
  *(uint4*)(dst + t * 8) = *(const uint4*)(src + t * 8);
}

// ------------------- LDS-staged WMMA GEMM kernels (bf16) --------------------
// WG = 8 waves (4x2); tile 128x64; per wave 32x32 = 2x2 frags; K-step 64,
// double-buffered async-to-LDS staging.

__device__ __forceinline__ void stage_tile(const u16* gbase, int ld, int rows,
                                           int row0, int k0, u16* lbase,
                                           int chunks_per_thread, int tid) {
  // tile is rows x 64 bf16 = rows*8 16B-chunks, row-major ld 64 in LDS
  const u16* g0 = gbase + (size_t)row0 * ld + k0;
#pragma unroll
  for (int i = 0; i < 4; i++) {
    if (i >= chunks_per_thread) break;
    int c = tid * chunks_per_thread + i;
    int row = c >> 3, ko = (c & 7) * 8;
    async_cp16(g0 + (size_t)row * ld + ko, lbase + row * 64 + ko);
  }
}

__global__ __launch_bounds__(256)
void k_gateup_lds(const u16* __restrict__ X, const u16* __restrict__ Wg,
                  const u16* __restrict__ Wu, u16* __restrict__ Aout,
                  int Kdim, int Ntot, size_t sX, size_t sW, size_t sA) {
  __shared__ u16 smem[2 * 128 * 64 + 2 * 64 * 64 + 2 * 64 * 64];  // 64 KB
  u16* bufA = smem;
  u16* bufG = smem + 2 * 128 * 64;
  u16* bufU = smem + 2 * 128 * 64 + 2 * 64 * 64;

  int g = blockIdx.z;
  int e = g % Ee;
  const u16* Xg  = X  + (size_t)g * sX;
  const u16* Wgp = Wg + (size_t)e * sW;
  const u16* Wup = Wu + (size_t)e * sW;
  u16* Ag = Aout + (size_t)g * sA;

  int tid  = threadIdx.x;
  int lane = tid & 31, wid = tid >> 5;
  int wm = wid >> 1, wn = wid & 1;
  int rowB = blockIdx.y * 128;         // block row origin
  int colB = blockIdx.x * 64;          // block col origin

  const v8f vz = {0.f, 0.f, 0.f, 0.f, 0.f, 0.f, 0.f, 0.f};
  v8f cg[4], cu[4];
#pragma unroll
  for (int i = 0; i < 4; i++) { cg[i] = vz; cu[i] = vz; }

  auto stage = [&](int b, int k0) {
    stage_tile(Xg,  Kdim, 128, rowB, k0, bufA + b * 128 * 64, 4, tid);
    stage_tile(Wgp, Kdim, 64,  colB, k0, bufG + b * 64 * 64,  2, tid);
    stage_tile(Wup, Kdim, 64,  colB, k0, bufU + b * 64 * 64,  2, tid);
  };

  int steps = Kdim >> 6;
  stage(0, 0);
  for (int it = 0; it < steps; ++it) {
    int bi = it & 1;
    if (it + 1 < steps) { stage(bi ^ 1, (it + 1) << 6); wait_async<8>(); }
    else                { wait_async<0>(); }
    __syncthreads();
    const u16* sA = bufA + bi * 128 * 64;
    const u16* sG = bufG + bi * 64 * 64;
    const u16* sU = bufU + bi * 64 * 64;
#pragma unroll
    for (int ks = 0; ks < 64; ks += 32) {
      v16bf a0  = load_fragA(sA, 64, wm * 32,      ks, lane);
      v16bf a1  = load_fragA(sA, 64, wm * 32 + 16, ks, lane);
      v16bf bg0 = load_fragB_bf16(sG, 64, wn * 32,      ks, lane);
      v16bf bg1 = load_fragB_bf16(sG, 64, wn * 32 + 16, ks, lane);
      v16bf bu0 = load_fragB_bf16(sU, 64, wn * 32,      ks, lane);
      v16bf bu1 = load_fragB_bf16(sU, 64, wn * 32 + 16, ks, lane);
      cg[0] = wmma_bf16(a0, bg0, cg[0]);
      cg[1] = wmma_bf16(a0, bg1, cg[1]);
      cg[2] = wmma_bf16(a1, bg0, cg[2]);
      cg[3] = wmma_bf16(a1, bg1, cg[3]);
      cu[0] = wmma_bf16(a0, bu0, cu[0]);
      cu[1] = wmma_bf16(a0, bu1, cu[1]);
      cu[2] = wmma_bf16(a1, bu0, cu[2]);
      cu[3] = wmma_bf16(a1, bu1, cu[3]);
    }
    __syncthreads();
  }

  int n = lane & 15, hf = lane >> 4;
#pragma unroll
  for (int mi = 0; mi < 2; mi++)
#pragma unroll
    for (int ni = 0; ni < 2; ni++) {
      v8f G = cg[mi * 2 + ni], U = cu[mi * 2 + ni];
#pragma unroll
      for (int r = 0; r < 8; r++) {
        int row = rowB + wm * 32 + mi * 16 + hf * 8 + r;
        int col = colB + wn * 32 + ni * 16 + n;
        float val = gelu_exact(G[r]) * U[r];
        Ag[(size_t)row * Ntot + col] = f2bf(val);
      }
    }
}

// MODE 0: shared-expert down-proj, plain store out[row*H + col] (first writer)
// MODE 1: routed down-proj, score-weighted transposed scatter-add
template <int MODE>
__global__ __launch_bounds__(256)
void k_down_lds(const u16* __restrict__ X, const u16* __restrict__ W,
                float* __restrict__ out, int Kdim, size_t sX, size_t sW,
                const int* __restrict__ idx, const float* __restrict__ score) {
  __shared__ u16 smem[2 * 128 * 64 + 2 * 64 * 64];                // 48 KB
  u16* bufA = smem;
  u16* bufW = smem + 2 * 128 * 64;

  int g = blockIdx.z;
  int e = g % Ee;
  const u16* Xg = X + (size_t)g * sX;
  const u16* Wp = W + (size_t)e * sW;

  int tid  = threadIdx.x;
  int lane = tid & 31, wid = tid >> 5;
  int wm = wid >> 1, wn = wid & 1;
  int rowB = blockIdx.y * 128;
  int colB = blockIdx.x * 64;

  const v8f vz = {0.f, 0.f, 0.f, 0.f, 0.f, 0.f, 0.f, 0.f};
  v8f c[4];
#pragma unroll
  for (int i = 0; i < 4; i++) c[i] = vz;

  auto stage = [&](int b, int k0) {
    stage_tile(Xg, Kdim, 128, rowB, k0, bufA + b * 128 * 64, 4, tid);
    stage_tile(Wp, Kdim, 64,  colB, k0, bufW + b * 64 * 64,  2, tid);
  };

  int steps = Kdim >> 6;
  stage(0, 0);
  for (int it = 0; it < steps; ++it) {
    int bi = it & 1;
    if (it + 1 < steps) { stage(bi ^ 1, (it + 1) << 6); wait_async<6>(); }
    else                { wait_async<0>(); }
    __syncthreads();
    const u16* sA = bufA + bi * 128 * 64;
    const u16* sW2 = bufW + bi * 64 * 64;
#pragma unroll
    for (int ks = 0; ks < 64; ks += 32) {
      v16bf a0 = load_fragA(sA, 64, wm * 32,      ks, lane);
      v16bf a1 = load_fragA(sA, 64, wm * 32 + 16, ks, lane);
      v16bf b0 = load_fragB_bf16(sW2, 64, wn * 32,      ks, lane);
      v16bf b1 = load_fragB_bf16(sW2, 64, wn * 32 + 16, ks, lane);
      c[0] = wmma_bf16(a0, b0, c[0]);
      c[1] = wmma_bf16(a0, b1, c[1]);
      c[2] = wmma_bf16(a1, b0, c[2]);
      c[3] = wmma_bf16(a1, b1, c[3]);
    }
    __syncthreads();
  }

  int n = lane & 15, hf = lane >> 4;
#pragma unroll
  for (int mi = 0; mi < 2; mi++)
#pragma unroll
    for (int ni = 0; ni < 2; ni++) {
      v8f C = c[mi * 2 + ni];
#pragma unroll
      for (int r = 0; r < 8; r++) {
        int m   = rowB + wm * 32 + mi * 16 + hf * 8 + r;
        int col = colB + wn * 32 + ni * 16 + n;
        if constexpr (MODE == 0) {
          out[(size_t)m * Hh + col] = C[r];
        } else {
          int   tok = idx[g * CAPc + m];
          float w   = score[g * CAPc + m];
          int   b   = g / Ee;
          atomicAdd(out + (size_t)b * Ss * Hh + (size_t)col * Ss + tok,
                    C[r] * w);
        }
      }
    }
}

// -------------- fallback GEMMs (fp32 weights, direct global) ----------------

template <bool WF32>
__global__ __launch_bounds__(256)
void k_gateup(const u16* __restrict__ X, const void* __restrict__ Wg,
              const void* __restrict__ Wu, u16* __restrict__ Aout,
              int Kdim, int Ntot, size_t sX, size_t sW, size_t sA) {
  int g = blockIdx.z;
  int e = g % Ee;
  const u16* Xg = X + (size_t)g * sX;
  const void* Wgp;
  const void* Wup;
  if constexpr (WF32) {
    Wgp = (const void*)((const float*)Wg + (size_t)e * sW);
    Wup = (const void*)((const float*)Wu + (size_t)e * sW);
  } else {
    Wgp = (const void*)((const u16*)Wg + (size_t)e * sW);
    Wup = (const void*)((const u16*)Wu + (size_t)e * sW);
  }
  u16* Ag = Aout + (size_t)g * sA;

  int lane = threadIdx.x & 31, wid = threadIdx.x >> 5;
  int wm = wid >> 1, wn = wid & 1;
  int row0 = blockIdx.y * 128 + wm * 32;
  int col0 = blockIdx.x * 64 + wn * 32;

  const v8f vz = {0.f, 0.f, 0.f, 0.f, 0.f, 0.f, 0.f, 0.f};
  v8f cg[4], cu[4];
#pragma unroll
  for (int i = 0; i < 4; i++) { cg[i] = vz; cu[i] = vz; }

  for (int k0 = 0; k0 < Kdim; k0 += 32) {
    __builtin_prefetch(Xg + (size_t)(row0 + (lane & 15)) * Kdim + k0 + 32, 0, 0);
    v16bf a0  = load_fragA(Xg, Kdim, row0,      k0, lane);
    v16bf a1  = load_fragA(Xg, Kdim, row0 + 16, k0, lane);
    v16bf bg0 = load_fragW<WF32>(Wgp, Kdim, col0,      k0, lane);
    v16bf bg1 = load_fragW<WF32>(Wgp, Kdim, col0 + 16, k0, lane);
    v16bf bu0 = load_fragW<WF32>(Wup, Kdim, col0,      k0, lane);
    v16bf bu1 = load_fragW<WF32>(Wup, Kdim, col0 + 16, k0, lane);
    cg[0] = wmma_bf16(a0, bg0, cg[0]);
    cg[1] = wmma_bf16(a0, bg1, cg[1]);
    cg[2] = wmma_bf16(a1, bg0, cg[2]);
    cg[3] = wmma_bf16(a1, bg1, cg[3]);
    cu[0] = wmma_bf16(a0, bu0, cu[0]);
    cu[1] = wmma_bf16(a0, bu1, cu[1]);
    cu[2] = wmma_bf16(a1, bu0, cu[2]);
    cu[3] = wmma_bf16(a1, bu1, cu[3]);
  }

  int n = lane & 15, hf = lane >> 4;
#pragma unroll
  for (int mi = 0; mi < 2; mi++)
#pragma unroll
    for (int ni = 0; ni < 2; ni++) {
      v8f G = cg[mi * 2 + ni], U = cu[mi * 2 + ni];
#pragma unroll
      for (int r = 0; r < 8; r++) {
        int row = row0 + mi * 16 + hf * 8 + r;
        int col = col0 + ni * 16 + n;
        float val = gelu_exact(G[r]) * U[r];
        Ag[(size_t)row * Ntot + col] = f2bf(val);
      }
    }
}

template <bool WF32, int MODE>
__global__ __launch_bounds__(256)
void k_down(const u16* __restrict__ X, const void* __restrict__ W,
            float* __restrict__ out, int Kdim, size_t sX, size_t sW,
            const int* __restrict__ idx, const float* __restrict__ score) {
  int g = blockIdx.z;
  int e = g % Ee;
  const u16* Xg = X + (size_t)g * sX;
  const void* Wp;
  if constexpr (WF32)
    Wp = (const void*)((const float*)W + (size_t)e * sW);
  else
    Wp = (const void*)((const u16*)W + (size_t)e * sW);

  int lane = threadIdx.x & 31, wid = threadIdx.x >> 5;
  int wm = wid >> 1, wn = wid & 1;
  int row0 = blockIdx.y * 128 + wm * 32;
  int col0 = blockIdx.x * 64 + wn * 32;

  const v8f vz = {0.f, 0.f, 0.f, 0.f, 0.f, 0.f, 0.f, 0.f};
  v8f c[4];
#pragma unroll
  for (int i = 0; i < 4; i++) c[i] = vz;

  for (int k0 = 0; k0 < Kdim; k0 += 32) {
    __builtin_prefetch(Xg + (size_t)(row0 + (lane & 15)) * Kdim + k0 + 32, 0, 0);
    v16bf a0 = load_fragA(Xg, Kdim, row0,      k0, lane);
    v16bf a1 = load_fragA(Xg, Kdim, row0 + 16, k0, lane);
    v16bf b0 = load_fragW<WF32>(Wp, Kdim, col0,      k0, lane);
    v16bf b1 = load_fragW<WF32>(Wp, Kdim, col0 + 16, k0, lane);
    c[0] = wmma_bf16(a0, b0, c[0]);
    c[1] = wmma_bf16(a0, b1, c[1]);
    c[2] = wmma_bf16(a1, b0, c[2]);
    c[3] = wmma_bf16(a1, b1, c[3]);
  }

  int n = lane & 15, hf = lane >> 4;
#pragma unroll
  for (int mi = 0; mi < 2; mi++)
#pragma unroll
    for (int ni = 0; ni < 2; ni++) {
      v8f C = c[mi * 2 + ni];
#pragma unroll
      for (int r = 0; r < 8; r++) {
        int m   = row0 + mi * 16 + hf * 8 + r;
        int col = col0 + ni * 16 + n;
        if constexpr (MODE == 0) {
          out[(size_t)m * Hh + col] = C[r];
        } else {
          int   tok = idx[g * CAPc + m];
          float w   = score[g * CAPc + m];
          int   b   = g / Ee;
          atomicAdd(out + (size_t)b * Ss * Hh + (size_t)col * Ss + tok,
                    C[r] * w);
        }
      }
    }
}

// --------------------------------- launcher ---------------------------------

extern "C" void kernel_launch(void* const* d_in, const int* in_sizes, int n_in,
                              void* d_out, int out_size, void* d_ws, size_t ws_size,
                              hipStream_t stream) {
  (void)in_sizes; (void)n_in; (void)out_size;

  const float* hs        = (const float*)d_in[0];
  const float* gate_w    = (const float*)d_in[1];
  const float* gate_proj = (const float*)d_in[2];
  const float* up_proj   = (const float*)d_in[3];
  const float* down_proj = (const float*)d_in[4];
  const float* s_gate    = (const float*)d_in[5];
  const float* s_up      = (const float*)d_in[6];
  const float* s_down    = (const float*)d_in[7];
  float* out = (float*)d_out;

  char* ws = (char*)d_ws;
  size_t off = 0;
  auto take = [&](size_t bytes) -> char* {
    char* p = ws + off;
    off = (off + bytes + 255) & ~(size_t)255;
    return p;
  };
  float* aff   = (float*)take((size_t)Bb * Ee * Ss * 4);
  int*   idx   = (int*)  take((size_t)Bb * Ee * CAPc * 4);
  float* score = (float*)take((size_t)Bb * Ee * CAPc * 4);
  u16*   Xs    = (u16*)  take((size_t)Bb * Ss * Hh * 2);          // 16 MB
  u16*   Xr    = (u16*)  take((size_t)Bb * Ee * CAPc * Hh * 2);   // 32 MB
  u16*   ACT   = (u16*)  take((size_t)Bb * Ee * CAPc * Ii * 2);   // 64 MB (reused)
  u16*   sgb   = (u16*)  take((size_t)Ish * Hh * 2);
  u16*   sub   = (u16*)  take((size_t)Ish * Hh * 2);
  u16*   sdb   = (u16*)  take((size_t)Hh * Ish * 2);
  // optional pre-converted routed weights (bf16) for the async/LDS fast path
  u16*   Wgb   = (u16*)  take((size_t)Ee * Ii * Hh * 2);          // 32 MB
  u16*   Wub   = (u16*)  take((size_t)Ee * Ii * Hh * 2);          // 32 MB
  u16*   Wdb   = (u16*)  take((size_t)Ee * Hh * Ii * 2);          // 32 MB
  bool full = (ws_size >= off);

  auto cvt = [&](const float* s, u16* d, int n) {
    int grid = (n / 4 + 255) / 256;
    k_cvt<<<grid, 256, 0, stream>>>(s, d, n);
  };
  cvt(hs,     Xs,  Bb * Ss * Hh);
  cvt(s_gate, sgb, Ish * Hh);
  cvt(s_up,   sub, Ish * Hh);
  cvt(s_down, sdb, Hh * Ish);
  if (full) {
    cvt(gate_proj, Wgb, Ee * Ii * Hh);
    cvt(up_proj,   Wub, Ee * Ii * Hh);
    cvt(down_proj, Wdb, Ee * Hh * Ii);
  }

  // gating + expert-choice top-k
  k_gating<<<(Bb * Ss) / 8, 256, 0, stream>>>(hs, gate_w, aff);
  k_topk<<<Bb * Ee, 1024, 0, stream>>>(aff, idx, score);

  // shared expert (async/LDS WMMA path; writes out first, plain store)
  k_gateup_lds<<<dim3(Ii / 64, (Bb * Ss) / 128, 1), 256, 0, stream>>>(
      Xs, sgb, sub, ACT, /*K=*/Hh, /*Ntot=*/Ish, 0, 0, 0);
  k_down_lds<0><<<dim3(Hh / 64, (Bb * Ss) / 128, 1), 256, 0, stream>>>(
      ACT, sdb, out, /*K=*/Ish, 0, 0, nullptr, nullptr);

  // routed experts: grouped over (b,e); transposed weighted scatter-add
  k_gather<<<Bb * Ee * CAPc, 128, 0, stream>>>(Xs, idx, Xr);
  if (full) {
    k_gateup_lds<<<dim3(Ii / 64, CAPc / 128, Bb * Ee), 256, 0, stream>>>(
        Xr, Wgb, Wub, ACT, Hh, Ii,
        (size_t)CAPc * Hh, (size_t)Ii * Hh, (size_t)CAPc * Ii);
    k_down_lds<1><<<dim3(Hh / 64, CAPc / 128, Bb * Ee), 256, 0, stream>>>(
        ACT, Wdb, out, Ii, (size_t)CAPc * Ii, (size_t)Hh * Ii, idx, score);
  } else {
    k_gateup<true><<<dim3(Ii / 64, CAPc / 128, Bb * Ee), 256, 0, stream>>>(
        Xr, gate_proj, up_proj, ACT, Hh, Ii,
        (size_t)CAPc * Hh, (size_t)Ii * Hh, (size_t)CAPc * Ii);
    k_down<true, 1><<<dim3(Hh / 64, CAPc / 128, Bb * Ee), 256, 0, stream>>>(
        ACT, down_proj, out, Ii, (size_t)CAPc * Ii, (size_t)Hh * Ii,
        idx, score);
  }
}
